// ChewyEncoder_81157702025451
// MI455X (gfx1250) — matrix operand order repaired and checked by reading, not compiled, not run on hardware
//
#include <hip/hip_runtime.h>
#include <hip/hip_bf16.h>

typedef __attribute__((ext_vector_type(16))) _Float16 v16h;
typedef __attribute__((ext_vector_type(8)))  _Float16 v8h;
typedef __attribute__((ext_vector_type(8)))  float    v8f;

#define KEIG 8
#define HIDN 64
#define PEW  32
#define HC   64
#define NEG_SLOPE 0.2f

// ---------------------------------------------------------------------------
// WMMA fragment index helpers (CDNA5 ISA 7.12.2, wave32)
// A (16-bit, 16x32): lane L holds row M=L&15.
//   K(lane,i) = kbase + (i<8 ? i : 16+(i-8)) + ((L&16)?8:0)   (contiguous spans)
// B (16-bit, 32x16): lane L holds col N=L&15; K same as A: (i<8 ? i : 16+(i-8)) + ((L&16)?8:0)
// C/D (f32, 16x16): element v -> row M = v + ((L&16)?8:0), col N = L&15
// ---------------------------------------------------------------------------
__device__ __forceinline__ int a_k_of(int lane, int i) {
  int hi = (lane & 16) ? 8 : 0;
  return (i < 8 ? i : 8 + i) + hi;   // i<8 -> i ; i>=8 -> 16+(i-8)
}
__device__ __forceinline__ int b_k_of(int lane, int i) {
  return (i < 8 ? i : 8 + i) + ((lane & 16) ? 8 : 0);
}
__device__ __forceinline__ v8f wmma16(v16h a, v16h b, v8f c) {
  return __builtin_amdgcn_wmma_f32_16x16x32_f16(false, a, false, b, (short)0, c,
                                                false, false);
}

// A fragment from (optionally concatenated) row-major f32 matrices.
// Fast path: whole 32-wide K chunk inside one source -> 4x float4 loads.
__device__ __forceinline__ v16h load_a_frag(const float* __restrict__ A0, int K0,
                                            const float* __restrict__ A1, int K1,
                                            int grow, bool rowok, int kbase,
                                            int lane) {
  const float* src;
  int ld, k0;
  if (kbase < K0) { src = A0; ld = K0; k0 = kbase; }
  else            { src = A1; ld = K1; k0 = kbase - K0; }
  const int hi = (lane & 16) ? 8 : 0;
  v16h a;
  if (rowok && (k0 + 32 <= ld)) {
    const float* base = src + (size_t)grow * ld + k0 + hi;
    float4 l0 = *(const float4*)(base + 0);
    float4 l1 = *(const float4*)(base + 4);
    float4 h0 = *(const float4*)(base + 16);
    float4 h1 = *(const float4*)(base + 20);
    a[0] = (_Float16)l0.x;  a[1] = (_Float16)l0.y;
    a[2] = (_Float16)l0.z;  a[3] = (_Float16)l0.w;
    a[4] = (_Float16)l1.x;  a[5] = (_Float16)l1.y;
    a[6] = (_Float16)l1.z;  a[7] = (_Float16)l1.w;
    a[8] = (_Float16)h0.x;  a[9] = (_Float16)h0.y;
    a[10] = (_Float16)h0.z; a[11] = (_Float16)h0.w;
    a[12] = (_Float16)h1.x; a[13] = (_Float16)h1.y;
    a[14] = (_Float16)h1.z; a[15] = (_Float16)h1.w;
  } else {
#pragma unroll
    for (int i = 0; i < 16; i++) {
      int K = kbase + a_k_of(lane, i);
      float v = 0.f;
      if (rowok) {
        if (K < K0)                      v = A0[(size_t)grow * K0 + K];
        else if (K1 > 0 && K - K0 < K1)  v = A1[(size_t)grow * K1 + (K - K0)];
      }
      a[i] = (_Float16)v;
    }
  }
  return a;
}

// A fragment from an LDS activation tile (row stride 72 halves, 16B aligned)
__device__ __forceinline__ v16h load_a_lds(const _Float16* t, int row,
                                           int kbase, int lane) {
  const int hi = (lane & 16) ? 8 : 0;
  v8h lo = *(const v8h*)(t + row * 72 + kbase + hi);
  v8h hh = *(const v8h*)(t + row * 72 + kbase + 16 + hi);
  v16h a;
#pragma unroll
  for (int i = 0; i < 8; i++) { a[i] = lo[i]; a[8 + i] = hh[i]; }
  return a;
}

// Pre-swizzled B fragment: one contiguous 32B span per lane
__device__ __forceinline__ v16h load_b_swz(const _Float16* __restrict__ Wsw,
                                           int frag, int lane) {
  return *(const v16h*)(Wsw + (size_t)frag * 512 + lane * 16);
}

// order-preserving float<->uint map for atomicMax on signed floats
__device__ __forceinline__ unsigned fmap(float f) {
  unsigned u = __float_as_uint(f);
  return (u & 0x80000000u) ? ~u : (u | 0x80000000u);
}
__device__ __forceinline__ float funmap(unsigned u) {
  unsigned v = (u & 0x80000000u) ? (u & 0x7FFFFFFFu) : ~u;
  return __uint_as_float(v);
}
__device__ __forceinline__ float eluf(float x) {
  return x > 0.f ? x : (__expf(x) - 1.f);
}

// ---------------------------------------------------------------------------
// Weight conversion + swizzle into WMMA-B fragment order.
// out[((kc*NT+nt)*32 + L)*16 + i] = f16(W[(kc*32 + b_k_of(L,i)) * Nout + nt*16 + (L&15)])
// ---------------------------------------------------------------------------
__global__ void cvt_swz_kernel(const float* __restrict__ W,
                               _Float16* __restrict__ out, int K, int Nout,
                               int total) {
  int t = blockIdx.x * blockDim.x + threadIdx.x;
  if (t >= total) return;
  int i = t & 15;
  int L = (t >> 4) & 31;
  int frag = t >> 9;
  int NT = Nout >> 4;
  int nt = frag % NT;
  int kc = frag / NT;
  int Kidx = kc * 32 + b_k_of(L, i);
  int col = nt * 16 + (L & 15);
  out[t] = (Kidx < K) ? (_Float16)W[(size_t)Kidx * Nout + col] : (_Float16)0.f;
}

// ---------------------------------------------------------------------------
// SignNet: pe = rho(relu-MLP(v) + relu-MLP(-v)).  One wave per 16 nodes.
// ---------------------------------------------------------------------------
__global__ void __launch_bounds__(32)
signnet_kernel(const float* __restrict__ eig,
               const _Float16* __restrict__ w1sw, const float* __restrict__ b1,
               const _Float16* __restrict__ w2sw, const float* __restrict__ b2,
               const _Float16* __restrict__ wrsw, const float* __restrict__ br,
               float* __restrict__ pe, int n) {
  __shared__ __align__(32) _Float16 tile[2][16 * 72];
  const int lane = threadIdx.x;
  const int base = blockIdx.x * 16;
  const int grow = base + (lane & 15);
  const int col0 = lane & 15;

  // ---- stage 1: [16x8] @ [8x64] (K padded to 32), both signs ----
  v16h ap = load_a_frag(eig, KEIG, nullptr, 0, grow, grow < n, 0, lane);
  v16h am;
#pragma unroll
  for (int i = 0; i < 16; i++) am[i] = -ap[i];
#pragma unroll
  for (int nt = 0; nt < 4; nt++) {
    int col = nt * 16 + col0;
    v16h b = load_b_swz(w1sw, nt, lane);
    v8f cz = {};
    v8f accp = wmma16(ap, b, cz);
    v8f accm = wmma16(am, b, cz);
    float bb = b1[col];
#pragma unroll
    for (int e = 0; e < 8; e++) {
      int r = e + ((lane & 16) ? 8 : 0);
      float fp = accp[e] + bb; fp = fp > 0.f ? fp : 0.f;
      float fm = accm[e] + bb; fm = fm > 0.f ? fm : 0.f;
      tile[0][r * 72 + col] = (_Float16)fp;
      tile[1][r * 72 + col] = (_Float16)fm;
    }
  }
  __syncthreads();

  // ---- stage 2: [16x64] @ [64x64], relu, sum over signs ----
  v8f ssum[4] = {};
  for (int sgn = 0; sgn < 2; sgn++) {
    v8f acc[4] = {};
#pragma unroll
    for (int kc = 0; kc < 2; kc++) {
      v16h a = load_a_lds(tile[sgn], lane & 15, kc * 32, lane);
#pragma unroll
      for (int nt = 0; nt < 4; nt++)
        acc[nt] = wmma16(a, load_b_swz(w2sw, kc * 4 + nt, lane), acc[nt]);
    }
#pragma unroll
    for (int nt = 0; nt < 4; nt++) {
      float bb = b2[nt * 16 + col0];
#pragma unroll
      for (int e = 0; e < 8; e++) {
        float f = acc[nt][e] + bb;
        ssum[nt][e] += (f > 0.f ? f : 0.f);
      }
    }
  }
  __syncthreads();
#pragma unroll
  for (int nt = 0; nt < 4; nt++) {
    int col = nt * 16 + col0;
#pragma unroll
    for (int e = 0; e < 8; e++) {
      int r = e + ((lane & 16) ? 8 : 0);
      tile[0][r * 72 + col] = (_Float16)ssum[nt][e];
    }
  }
  __syncthreads();

  // ---- stage 3: rho  [16x64] @ [64x32] ----
  v8f accr[2] = {};
#pragma unroll
  for (int kc = 0; kc < 2; kc++) {
    v16h a = load_a_lds(tile[0], lane & 15, kc * 32, lane);
#pragma unroll
    for (int nt = 0; nt < 2; nt++)
      accr[nt] = wmma16(a, load_b_swz(wrsw, kc * 2 + nt, lane), accr[nt]);
  }
#pragma unroll
  for (int nt = 0; nt < 2; nt++) {
    int col = nt * 16 + col0;
    float bb = br[col];
#pragma unroll
    for (int e = 0; e < 8; e++) {
      int r = base + e + ((lane & 16) ? 8 : 0);
      if (r < n) pe[(size_t)r * PEW + col] = accr[nt][e] + bb;
    }
  }
}

// ---------------------------------------------------------------------------
// Node projection: out[N,64] = concat(A0[N,K0], A1[N,K1]) @ W[K0+K1,64]
// One wave per 16 rows; W pre-swizzled (NT=4).  GAT bias applied later.
// ---------------------------------------------------------------------------
__global__ void __launch_bounds__(32)
lin64_kernel(const float* __restrict__ A0, int K0,
             const float* __restrict__ A1, int K1,
             const _Float16* __restrict__ Wsw, float* __restrict__ out, int n) {
  const int lane = threadIdx.x;
  const int base = blockIdx.x * 16;
  const int grow = base + (lane & 15);
  const int col0 = lane & 15;
  const int Kt = K0 + K1;

  v8f acc[4] = {};
  for (int kc = 0; kc * 32 < Kt; kc++) {
    v16h a = load_a_frag(A0, K0, A1, K1, grow, grow < n, kc * 32, lane);
#pragma unroll
    for (int nt = 0; nt < 4; nt++)
      acc[nt] = wmma16(a, load_b_swz(Wsw, kc * 4 + nt, lane), acc[nt]);
  }
#pragma unroll
  for (int nt = 0; nt < 4; nt++) {
    int col = nt * 16 + col0;
#pragma unroll
    for (int e = 0; e < 8; e++) {
      int r = base + e + ((lane & 16) ? 8 : 0);
      if (r < n) out[(size_t)r * HC + col] = acc[nt][e];
    }
  }
}

// ---------------------------------------------------------------------------
// Per-node attention logits: alpha_s[n,h], alpha_d[n,h]
// ---------------------------------------------------------------------------
__global__ void alpha_kernel(const float* __restrict__ h,
                             const float* __restrict__ asrc,
                             const float* __restrict__ adst,
                             float* __restrict__ als, float* __restrict__ ald,
                             int n) {
  int tid = blockIdx.x * blockDim.x + threadIdx.x;
  if (tid >= n * 2) return;
  int node = tid >> 1, head = tid & 1;
  float s = 0.f, d = 0.f;
#pragma unroll
  for (int c4 = 0; c4 < 32; c4 += 4) {
    float4 hv = *(const float4*)(h + (size_t)node * HC + head * 32 + c4);
    float4 as = *(const float4*)(asrc + head * 32 + c4);
    float4 ad = *(const float4*)(adst + head * 32 + c4);
    s += hv.x * as.x + hv.y * as.y + hv.z * as.z + hv.w * as.w;
    d += hv.x * ad.x + hv.y * ad.y + hv.z * ad.z + hv.w * ad.w;
  }
  als[tid] = s;
  ald[tid] = d;
}

// ---------------------------------------------------------------------------
// Edge pass 1: segment max of leaky-relu logits (atomicMax on mapped uint)
// ---------------------------------------------------------------------------
__global__ void edge_max_kernel(const int* __restrict__ src,
                                const int* __restrict__ dst,
                                const float* __restrict__ als,
                                const float* __restrict__ ald,
                                unsigned* __restrict__ emax, int E) {
  int e = blockIdx.x * blockDim.x + threadIdx.x;
  if (e >= E) return;
  int s = src[e], d = dst[e];
#pragma unroll
  for (int h = 0; h < 2; h++) {
    float v = als[s * 2 + h] + ald[d * 2 + h];
    v = v > 0.f ? v : NEG_SLOPE * v;
    atomicMax(&emax[d * 2 + h], fmap(v));
  }
}

// ---------------------------------------------------------------------------
// Edge pass 2: e_exp and segment denom
// ---------------------------------------------------------------------------
__global__ void edge_exp_kernel(const int* __restrict__ src,
                                const int* __restrict__ dst,
                                const float* __restrict__ als,
                                const float* __restrict__ ald,
                                const unsigned* __restrict__ emax,
                                float* __restrict__ eexp,
                                float* __restrict__ denom, int E) {
  int e = blockIdx.x * blockDim.x + threadIdx.x;
  if (e >= E) return;
  int s = src[e], d = dst[e];
#pragma unroll
  for (int h = 0; h < 2; h++) {
    float v = als[s * 2 + h] + ald[d * 2 + h];
    v = v > 0.f ? v : NEG_SLOPE * v;
    float ex = __expf(v - funmap(emax[d * 2 + h]));
    eexp[(size_t)e * 2 + h] = ex;
    atomicAdd(&denom[d * 2 + h], ex);
  }
}

// ---------------------------------------------------------------------------
// Edge pass 3: weighted message scatter-add.  thread = (edge, 4 channels)
// ---------------------------------------------------------------------------
__global__ void edge_msg_kernel(const int* __restrict__ src,
                                const int* __restrict__ dst,
                                const float* __restrict__ h,
                                const float* __restrict__ eexp,
                                const float* __restrict__ denom,
                                float* __restrict__ agg, long long total) {
  long long tid = (long long)blockIdx.x * blockDim.x + threadIdx.x;
  if (tid >= total) return;
  int e = (int)(tid >> 4);
  int c = (int)(tid & 15) * 4;
  int s = src[e], d = dst[e];
  int head = c >> 5;
  float w = eexp[(size_t)e * 2 + head] / (denom[d * 2 + head] + 1e-16f);
  float4 hv = *(const float4*)(h + (size_t)s * HC + c);
  float* ap = agg + (size_t)d * HC + c;
  atomicAdd(ap + 0, hv.x * w);
  atomicAdd(ap + 1, hv.y * w);
  atomicAdd(ap + 2, hv.z * w);
  atomicAdd(ap + 3, hv.w * w);
}

// ---------------------------------------------------------------------------
// Node finalize: out = elu(agg + bias)
// ---------------------------------------------------------------------------
__global__ void node_act_kernel(const float* __restrict__ agg,
                                const float* __restrict__ bias,
                                float* __restrict__ out, int total) {
  int tid = blockIdx.x * blockDim.x + threadIdx.x;
  if (tid >= total) return;
  int c = tid & 63;
  out[tid] = eluf(agg[tid] + bias[c]);
}

// ---------------------------------------------------------------------------
// Final fused ELU + LayerNorm, one wave (32 lanes) per node, shfl reductions
// ---------------------------------------------------------------------------
__global__ void __launch_bounds__(32)
final_ln_kernel(const float* __restrict__ agg, const float* __restrict__ bias,
                const float* __restrict__ g, const float* __restrict__ bl,
                float* __restrict__ out, int n) {
  int node = blockIdx.x;
  int lane = threadIdx.x;
  if (node >= n) return;
  float v0 = eluf(agg[(size_t)node * HC + lane] + bias[lane]);
  float v1 = eluf(agg[(size_t)node * HC + 32 + lane] + bias[32 + lane]);
  float s = v0 + v1;
#pragma unroll
  for (int m = 16; m >= 1; m >>= 1) s += __shfl_xor(s, m, 32);
  float mu = s * (1.f / 64.f);
  float d0 = v0 - mu, d1 = v1 - mu;
  float q = d0 * d0 + d1 * d1;
#pragma unroll
  for (int m = 16; m >= 1; m >>= 1) q += __shfl_xor(q, m, 32);
  float inv = rsqrtf(q * (1.f / 64.f) + 1e-5f);
  out[(size_t)node * HC + lane] = d0 * inv * g[lane] + bl[lane];
  out[(size_t)node * HC + 32 + lane] = d1 * inv * g[32 + lane] + bl[32 + lane];
}

// ---------------------------------------------------------------------------
extern "C" void kernel_launch(void* const* d_in, const int* in_sizes, int n_in,
                              void* d_out, int out_size, void* d_ws,
                              size_t ws_size, hipStream_t stream) {
  const float* x      = (const float*)d_in[0];
  const float* eig    = (const float*)d_in[1];
  const int*   ei     = (const int*)d_in[2];
  const float* phi_w1 = (const float*)d_in[3];
  const float* phi_b1 = (const float*)d_in[4];
  const float* phi_w2 = (const float*)d_in[5];
  const float* phi_b2 = (const float*)d_in[6];
  const float* rho_w  = (const float*)d_in[7];
  const float* rho_b  = (const float*)d_in[8];
  const float* W0     = (const float*)d_in[9];
  const float* asrc0  = (const float*)d_in[10];
  const float* adst0  = (const float*)d_in[11];
  const float* b0     = (const float*)d_in[12];
  const float* W1     = (const float*)d_in[13];
  const float* asrc1  = (const float*)d_in[14];
  const float* adst1  = (const float*)d_in[15];
  const float* b1     = (const float*)d_in[16];
  const float* ln_g   = (const float*)d_in[17];
  const float* ln_b   = (const float*)d_in[18];
  float* outp = (float*)d_out;

  const int N = in_sizes[0] / 128;
  const int E = in_sizes[2] / 2;
  const int* src = ei;
  const int* dst = ei + E;

  // ---- workspace carve-out ----
  char* w = (char*)d_ws;
  size_t off = 0;
  auto carve = [&](size_t bytes) -> char* {
    char* p = w + off;
    off += (bytes + 255) & ~(size_t)255;
    return p;
  };
  float*    pe    = (float*)carve((size_t)N * PEW * 4);
  float*    hlin  = (float*)carve((size_t)N * HC * 4);   // proj out (both layers)
  float*    feat1 = (float*)carve((size_t)N * HC * 4);   // elu(agg0+b0)
  float*    agg   = (float*)carve((size_t)N * HC * 4);
  float*    als   = (float*)carve((size_t)N * 2 * 4);
  float*    ald   = (float*)carve((size_t)N * 2 * 4);
  unsigned* emax  = (unsigned*)carve((size_t)N * 2 * 4);
  float*    denom = (float*)carve((size_t)N * 2 * 4);
  float*    eexp  = (float*)carve((size_t)E * 2 * 4);
  // swizzled f16 weights: chunks(K) * (Nout/16) * 512 halves each
  const int szW0 = 5 * 4 * 512, szW1 = 2 * 4 * 512;
  const int szP1 = 1 * 4 * 512, szP2 = 2 * 4 * 512, szRW = 2 * 2 * 512;
  _Float16* w0sw = (_Float16*)carve((size_t)szW0 * 2);
  _Float16* w1sw = (_Float16*)carve((size_t)szW1 * 2);
  _Float16* p1sw = (_Float16*)carve((size_t)szP1 * 2);
  _Float16* p2sw = (_Float16*)carve((size_t)szP2 * 2);
  _Float16* rwsw = (_Float16*)carve((size_t)szRW * 2);
  (void)ws_size; (void)n_in; (void)out_size;

  const int TB = 256;
  auto g1 = [&](long long t) { return (unsigned)((t + TB - 1) / TB); };

  // 1) weight conversion + B-fragment swizzle
  cvt_swz_kernel<<<g1(szW0), TB, 0, stream>>>(W0, w0sw, 160, HC, szW0);
  cvt_swz_kernel<<<g1(szW1), TB, 0, stream>>>(W1, w1sw, HC, HC, szW1);
  cvt_swz_kernel<<<g1(szP1), TB, 0, stream>>>(phi_w1, p1sw, KEIG, HIDN, szP1);
  cvt_swz_kernel<<<g1(szP2), TB, 0, stream>>>(phi_w2, p2sw, HIDN, HIDN, szP2);
  cvt_swz_kernel<<<g1(szRW), TB, 0, stream>>>(rho_w, rwsw, HIDN, PEW, szRW);

  const unsigned tiles = (unsigned)((N + 15) / 16);

  // 2) SignNet PE
  signnet_kernel<<<tiles, 32, 0, stream>>>(eig, p1sw, phi_b1, p2sw, phi_b2,
                                           rwsw, rho_b, pe, N);

  // ---- GAT layer 0 ----
  lin64_kernel<<<tiles, 32, 0, stream>>>(x, 128, pe, PEW, w0sw, hlin, N);
  alpha_kernel<<<g1((long long)N * 2), TB, 0, stream>>>(hlin, asrc0, adst0,
                                                        als, ald, N);
  hipMemsetAsync(emax, 0, (size_t)N * 2 * 4, stream);
  hipMemsetAsync(denom, 0, (size_t)N * 2 * 4, stream);
  hipMemsetAsync(agg, 0, (size_t)N * HC * 4, stream);
  edge_max_kernel<<<g1(E), TB, 0, stream>>>(src, dst, als, ald, emax, E);
  edge_exp_kernel<<<g1(E), TB, 0, stream>>>(src, dst, als, ald, emax, eexp,
                                            denom, E);
  edge_msg_kernel<<<g1((long long)E * 16), TB, 0, stream>>>(
      src, dst, hlin, eexp, denom, agg, (long long)E * 16);
  node_act_kernel<<<g1((long long)N * HC), TB, 0, stream>>>(agg, b0, feat1,
                                                            N * HC);

  // ---- GAT layer 1 ----
  lin64_kernel<<<tiles, 32, 0, stream>>>(feat1, HC, nullptr, 0, w1sw, hlin, N);
  alpha_kernel<<<g1((long long)N * 2), TB, 0, stream>>>(hlin, asrc1, adst1,
                                                        als, ald, N);
  hipMemsetAsync(emax, 0, (size_t)N * 2 * 4, stream);
  hipMemsetAsync(denom, 0, (size_t)N * 2 * 4, stream);
  hipMemsetAsync(agg, 0, (size_t)N * HC * 4, stream);
  edge_max_kernel<<<g1(E), TB, 0, stream>>>(src, dst, als, ald, emax, E);
  edge_exp_kernel<<<g1(E), TB, 0, stream>>>(src, dst, als, ald, emax, eexp,
                                            denom, E);
  edge_msg_kernel<<<g1((long long)E * 16), TB, 0, stream>>>(
      src, dst, hlin, eexp, denom, agg, (long long)E * 16);

  // ---- final ELU + LayerNorm ----
  final_ln_kernel<<<(unsigned)N, 32, 0, stream>>>(agg, b1, ln_g, ln_b, outp, N);
}
